// Contextual_Patches_Score_Module_66159676228281
// MI455X (gfx1250) — matrix operand order repaired and checked by reading, not compiled, not run on hardware
//
#include <hip/hip_runtime.h>
#include <hip/hip_bf16.h>

typedef __attribute__((ext_vector_type(16))) __bf16 v16bf;
typedef __attribute__((ext_vector_type(8)))  __bf16 v8bf;
typedef __attribute__((ext_vector_type(8)))  float  v8f;

#define HS    80        // subsampled spatial size (160/2)
#define NCH   64        // channels
#define NPOS  6400      // HS*HS positions
#define KDIM  576       // 64*3*3 patch length (18 * 32)
#define TILEP 128       // output tile rows (P / b-patches)
#define TILEQ 256       // output tile cols (Q / f-positions)
#define LDST  40        // padded LDS row stride in bf16 (80B, 16B-aligned, conflict-free)

// ---------------------------------------------------------------------------
// Kernel 1: strided patch extraction + L2 normalization of b-patches.
// One 256-thread block per (batch, position). Writes bf16 im2col matrices:
//   Am[n][p][k] = b-patch(p) / max(||b-patch(p)||, 1e-4)   (GEMM A, row-major K)
//   Bm[n][q][k] = f-patch(q)                               (GEMM B, row-major K)
// ---------------------------------------------------------------------------
__global__ __launch_bounds__(256) void prep_kernel(
    const float* __restrict__ f, const float* __restrict__ b,
    __bf16* __restrict__ Am, __bf16* __restrict__ Bm)
{
  __shared__ float red[256];
  __shared__ float scale_sh;

  const int p  = blockIdx.x;
  const int n  = blockIdx.y;
  const int py = p / HS, px = p % HS;
  const int t  = threadIdx.x;

  float bval[3] = {0.f, 0.f, 0.f};
  float fval[3] = {0.f, 0.f, 0.f};
  float ssq = 0.f;

  #pragma unroll
  for (int i = 0; i < 3; ++i) {
    const int k = t + 256 * i;
    if (k < KDIM) {
      const int c   = k / 9;
      const int rem = k % 9;
      const int ky  = rem / 3, kx = rem % 3;
      const int iy  = py + ky - 1, ix = px + kx - 1;       // SAME padding on 80x80 grid
      if (iy >= 0 && iy < HS && ix >= 0 && ix < HS) {
        const size_t g = (((size_t)(n * NCH + c)) * 160 + 2 * iy) * 160 + 2 * ix; // ::2 subsample
        bval[i] = b[g];
        fval[i] = f[g];
      }
      ssq += bval[i] * bval[i];
    }
  }

  red[t] = ssq;
  __syncthreads();
  for (int s = 128; s > 0; s >>= 1) {
    if (t < s) red[t] += red[t + s];
    __syncthreads();
  }
  if (t == 0) scale_sh = 1.0f / fmaxf(sqrtf(red[0]), 1e-4f);
  __syncthreads();
  const float scale = scale_sh;

  const size_t rowb = ((size_t)n * NPOS + p) * KDIM;
  #pragma unroll
  for (int i = 0; i < 3; ++i) {
    const int k = t + 256 * i;
    if (k < KDIM) {
      Am[rowb + k] = (__bf16)(bval[i] * scale);
      Bm[rowb + k] = (__bf16)(fval[i]);
    }
  }
}

// ---------------------------------------------------------------------------
// Kernel 2: Y[n,p,q] = sum_k Am[n,p,k] * Bm[n,q,k]  via v_wmma_f32_16x16x32_bf16
// 128x256 tile per block; 8 wave32 waves in a 2(P) x 4(Q) grid, each wave owns
// a 64x64 sub-tile = 4x4 accumulators of 16x16 f32 (16 WMMAs / 8 fragments per
// K-step -> 4x fragment reuse on both operands).
// ---------------------------------------------------------------------------
__global__ __launch_bounds__(256) void gemm_kernel(
    const __bf16* __restrict__ Am, const __bf16* __restrict__ Bm,
    float* __restrict__ out)
{
  __shared__ __bf16 As[TILEP][LDST];   // 128 x 40 bf16 = 10240 B
  __shared__ __bf16 Bs[TILEQ][LDST];   // 256 x 40 bf16 = 20480 B

  const int t    = threadIdx.x;
  const int n    = blockIdx.z;
  const int p0   = blockIdx.x * TILEP;
  const int q0   = blockIdx.y * TILEQ;

  const int wave = t >> 5;
  const int lane = t & 31;
  const int h    = lane >> 4;    // 0/1: lane half (ISA 7.12.2 layouts)
  const int r    = lane & 15;    // row (A: M, B/D: N)
  const int wp   = wave & 1;     // P-offset 64*wp
  const int wq   = wave >> 1;    // Q-offset 64*wq (0..3)

  // staging assignment: A tile -> thread covers row t>>1, 16-bf16 half (t&1)
  //                     B tile -> thread covers row t, all 32 bf16
  const int aRow = t >> 1;
  const int aCol = (t & 1) * 16;
  const __bf16* gA = Am + ((size_t)(n * NPOS + p0 + aRow)) * KDIM + aCol;
  const __bf16* gB = Bm + ((size_t)(n * NPOS + q0 + t)) * KDIM;

  const v8f zero = {0.f, 0.f, 0.f, 0.f, 0.f, 0.f, 0.f, 0.f};
  v8f acc[4][4];
  #pragma unroll
  for (int i = 0; i < 4; ++i)
    #pragma unroll
    for (int j = 0; j < 4; ++j) acc[i][j] = zero;

  for (int k0 = 0; k0 < KDIM; k0 += 32) {
    // stage 128x32 (A) + 256x32 (B) bf16 tiles: 6 x b128 global loads / thread
    const uint4 a0 = *(const uint4*)(gA + k0);
    const uint4 a1 = *(const uint4*)(gA + k0 + 8);
    const uint4 b0 = *(const uint4*)(gB + k0);
    const uint4 b1 = *(const uint4*)(gB + k0 + 8);
    const uint4 b2 = *(const uint4*)(gB + k0 + 16);
    const uint4 b3 = *(const uint4*)(gB + k0 + 24);
    __syncthreads();                       // previous compute done before overwrite
    *(uint4*)&As[aRow][aCol]     = a0;
    *(uint4*)&As[aRow][aCol + 8] = a1;
    *(uint4*)&Bs[t][0]  = b0;
    *(uint4*)&Bs[t][8]  = b1;
    *(uint4*)&Bs[t][16] = b2;
    *(uint4*)&Bs[t][24] = b3;
    __syncthreads();

    // A fragment: lane(m=r,h): elements j<8 -> K = j + 8h ; j>=8 -> K = j+8 + 8h
    v16bf afr[4];
    #pragma unroll
    for (int i = 0; i < 4; ++i) {
      const int row = 64 * wp + 16 * i + r;
      const v8bf lo = *(const v8bf*)&As[row][8 * h];
      const v8bf hi = *(const v8bf*)&As[row][16 + 8 * h];
      afr[i] = __builtin_shufflevector(lo, hi, 0,1,2,3,4,5,6,7,8,9,10,11,12,13,14,15);
    }
    // B fragment: lane(n=r,h): elements j<8 -> K = j + 8h ; j>=8 -> K = j+8 + 8h
    v16bf bfr[4];
    #pragma unroll
    for (int j = 0; j < 4; ++j) {
      const int row = 64 * wq + 16 * j + r;
      const v8bf lo = *(const v8bf*)&Bs[row][8 * h];
      const v8bf hi = *(const v8bf*)&Bs[row][16 + 8 * h];
      bfr[j] = __builtin_shufflevector(lo, hi, 0,1,2,3,4,5,6,7,8,9,10,11,12,13,14,15);
    }

    #pragma unroll
    for (int i = 0; i < 4; ++i)
      #pragma unroll
      for (int j = 0; j < 4; ++j)
        acc[i][j] = __builtin_amdgcn_wmma_f32_16x16x32_bf16(
            false, afr[i], false, bfr[j], (short)0, acc[i][j], false, false);
  }

  // D layout: VGPR rr -> row m = rr + 8h, lanes give col n = r
  #pragma unroll
  for (int i = 0; i < 4; ++i) {
    #pragma unroll
    for (int j = 0; j < 4; ++j) {
      const int pm = p0 + 64 * wp + 16 * i + 8 * h;
      const int qn = q0 + 64 * wq + 16 * j + r;
      float* op = out + ((size_t)(n * NPOS + pm)) * (size_t)NPOS + qn;
      #pragma unroll
      for (int rr = 0; rr < 8; ++rr)
        op[(size_t)rr * NPOS] = acc[i][j][rr];
    }
  }
}

extern "C" void kernel_launch(void* const* d_in, const int* in_sizes, int n_in,
                              void* d_out, int out_size, void* d_ws, size_t ws_size,
                              hipStream_t stream) {
  (void)in_sizes; (void)n_in; (void)out_size; (void)ws_size;
  const float* f = (const float*)d_in[0];
  const float* b = (const float*)d_in[1];
  float* out = (float*)d_out;

  // workspace: bf16 im2col matrices, 2 * 6400 * 576 each (14.75 MB each)
  __bf16* Am = (__bf16*)d_ws;
  __bf16* Bm = Am + (size_t)2 * NPOS * KDIM;

  dim3 gPrep(NPOS, 2);
  prep_kernel<<<gPrep, 256, 0, stream>>>(f, b, Am, Bm);

  dim3 gGemm(NPOS / TILEP, NPOS / TILEQ, 2);   // 50 x 25 x 2
  gemm_kernel<<<gGemm, 256, 0, stream>>>(Am, Bm, out);
}
